// Decoder_72928544686482
// MI455X (gfx1250) — matrix operand order repaired and checked by reading, not compiled, not run on hardware
//
#include <hip/hip_runtime.h>
#include <math.h>

// ---------------------------------------------------------------------------
// Types for CDNA5 WMMA (gfx1250, wave32)
// ---------------------------------------------------------------------------
typedef __bf16 bf16;
typedef bf16  v16bf __attribute__((ext_vector_type(16)));
typedef float v8f   __attribute__((ext_vector_type(8)));
typedef int   v4i   __attribute__((ext_vector_type(4)));

#define B_  64
#define T_  32
#define E_  256
#define U_  512
#define G4  2048   // 4*U
#define ROWS 2048  // B*T
#define S_  31     // outer scan length (T-1)
#define NWG 8      // persistent workgroups for the LSTM recurrence

__device__ inline float sigm(float x) { return 1.0f / (1.0f + expf(-x)); }

__device__ inline v8f wmma_bf16(v16bf a, v16bf b, v8f c) {
  return __builtin_amdgcn_wmma_f32_16x16x32_bf16(false, a, false, b, (short)0, c,
                                                 false, false);
}

// A-tile (16x32 bf16) from row-major [.., ld] matrix. ISA 16-bit A layout:
// lane l<16 holds row l, K = {k0..k0+7, k0+16..k0+23}; lane l>=16 shifts +8.
__device__ inline v16bf load_a_tile(const bf16* base, int row0, int ld, int k0) {
  int lane = threadIdx.x & 31;
  const bf16* p = base + (row0 + (lane & 15)) * ld + k0 + ((lane >> 4) << 3);
  union { v16bf v; v4i q[2]; } u;
  u.q[0] = *(const v4i*)(p);
  u.q[1] = *(const v4i*)(p + 16);
  return u.v;
}

// A-tile with all 16 rows identical (broadcast of one vector)
__device__ inline v16bf load_a_bcast(const bf16* rowptr, int k0) {
  int lane = threadIdx.x & 31;
  const bf16* p = rowptr + k0 + ((lane >> 4) << 3);
  union { v16bf v; v4i q[2]; } u;
  u.q[0] = *(const v4i*)(p);
  u.q[1] = *(const v4i*)(p + 16);
  return u.v;
}

// B-tile (32x16 bf16) from pre-packed layout: tile = 512 bf16, lane-contiguous.
__device__ inline v16bf load_b_tile(const bf16* packed, int tile) {
  int lane = threadIdx.x & 31;
  const bf16* p = packed + tile * 512 + lane * 16;
  union { v16bf v; v4i q[2]; } u;
  u.q[0] = *(const v4i*)(p);
  u.q[1] = *(const v4i*)(p + 8);
  return u.v;
}

// Device-wide barrier for the persistent LSTM kernel. Monotonic counter;
// agent-scope atomics so the spin bypasses the WGP$ (CDNA5 SCOPE rules).
__device__ inline void grid_barrier(int* ctr, int target) {
  __syncthreads();
  if (threadIdx.x == 0) {
    __threadfence();  // release our h/c stores to device scope
    __hip_atomic_fetch_add(ctr, 1, __ATOMIC_RELEASE, __HIP_MEMORY_SCOPE_AGENT);
    while (__hip_atomic_load(ctr, __ATOMIC_ACQUIRE, __HIP_MEMORY_SCOPE_AGENT) <
           target)
      __builtin_amdgcn_s_sleep(1);
  }
  __syncthreads();
  __threadfence();    // acquire: invalidate near caches before reading peers' h
}

// ---------------------------------------------------------------------------
// Weight packing: row-major f32 [K][N] -> bf16 WMMA-B tiles.
// Packed element e: tile = e>>9 (= kt*(N/16)+ntb), lane l=(e>>4)&31, j=e&15.
// Source: k = kt*32 + (l>=16 ? 8 : 0) + (j&7) + (j>=8 ? 16 : 0) ; n = ntb*16 + (l&15).
// ---------------------------------------------------------------------------
__global__ void pack_b_kernel(const float* __restrict__ src, bf16* __restrict__ dst,
                              int K, int N, int total) {
  int NT = N >> 4;
  for (int e = blockIdx.x * blockDim.x + threadIdx.x; e < total;
       e += gridDim.x * blockDim.x) {
    int j = e & 15, l = (e >> 4) & 31, tile = e >> 9;
    int ntb = tile % NT, kt = tile / NT;
    int k = kt * 32 + ((l >> 4) << 3) + (j & 7) + ((j >> 3) << 4);
    int n = ntb * 16 + (l & 15);
    dst[e] = (bf16)src[k * N + n];
  }
}

__global__ void f32_to_bf16_kernel(const float* __restrict__ s, bf16* __restrict__ d,
                                   int n) {
  for (int i = blockIdx.x * blockDim.x + threadIdx.x; i < n;
       i += gridDim.x * blockDim.x)
    d[i] = (bf16)s[i];
}

__global__ void init_carry_kernel(const float* __restrict__ h0,
                                  const float* __restrict__ c0,
                                  float* __restrict__ hc, float* __restrict__ cc,
                                  int n) {
  int i = blockIdx.x * blockDim.x + threadIdx.x;
  if (i < n) { hc[i] = h0[i]; cc[i] = c0[i]; }
}

// dec_in0 = emb[target] -> bf16 [ROWS][E]; one wave per row
__global__ __launch_bounds__(128) void embed_init_kernel(
    const int* __restrict__ target, const float* __restrict__ emb,
    bf16* __restrict__ decb) {
  int wave = threadIdx.x >> 5, lane = threadIdx.x & 31;
  int row = blockIdx.x * 4 + wave;
  int tok = target[row];
  const float* e = emb + (long)tok * E_;
  for (int j = lane; j < E_; j += 32) decb[row * E_ + j] = (bf16)e[j];
}

// ---------------------------------------------------------------------------
// GEMM: xg = dec_in(bf16) @ Wk(packed) + b -> f32 [ROWS][G4]
// one wave per 16x16 tile; tiles = 128 x 128. Also resets the LSTM barrier
// counter (runs immediately before each lstm_seq launch on the stream).
// ---------------------------------------------------------------------------
__global__ __launch_bounds__(128) void gemm_xg_kernel(
    const bf16* __restrict__ decb, const bf16* __restrict__ Wkp,
    const float* __restrict__ bias, float* __restrict__ xg, int* barrier_ctr) {
  if (blockIdx.x == 0 && threadIdx.x == 0) *barrier_ctr = 0;
  int wave = threadIdx.x >> 5, lane = threadIdx.x & 31;
  int tile = blockIdx.x * 4 + wave;
  int mt = tile >> 7, nt = tile & 127;
  v8f acc = {0.f, 0.f, 0.f, 0.f, 0.f, 0.f, 0.f, 0.f};
  for (int kt = 0; kt < 8; ++kt) {
    v16bf A = load_a_tile(decb, mt * 16, E_, kt * 32);
    v16bf Bm = load_b_tile(Wkp, kt * 128 + nt);
    acc = wmma_bf16(A, Bm, acc);
  }
  int col = nt * 16 + (lane & 15);
  float bv = bias[col];
  int roff = (lane >> 4) << 3;
  for (int v = 0; v < 8; ++v)
    xg[(mt * 16 + v + roff) * G4 + col] = acc[v] + bv;
}

// Z_enc = enc_out(bf16) @ Wgen[0:512](packed) + bgen -> f32 [ROWS][U]
__global__ __launch_bounds__(128) void gemm_zenc_kernel(
    const bf16* __restrict__ encb, const bf16* __restrict__ Wge,
    const float* __restrict__ bgen, float* __restrict__ Zenc) {
  int wave = threadIdx.x >> 5, lane = threadIdx.x & 31;
  int tile = blockIdx.x * 4 + wave;   // 128*32 tiles
  int mt = tile >> 5, nt = tile & 31;
  v8f acc = {0.f, 0.f, 0.f, 0.f, 0.f, 0.f, 0.f, 0.f};
  for (int kt = 0; kt < 16; ++kt) {
    v16bf A = load_a_tile(encb, mt * 16, U_, kt * 32);
    v16bf Bm = load_b_tile(Wge, kt * 32 + nt);
    acc = wmma_bf16(A, Bm, acc);
  }
  int col = nt * 16 + (lane & 15);
  float bv = bgen[col];
  int roff = (lane >> 4) << 3;
  for (int v = 0; v < 8; ++v)
    Zenc[(mt * 16 + v + roff) * U_ + col] = acc[v] + bv;
}

// p_gen[s] = sigmoid(Zenc + h0 @ Wgen_h + dec_in @ Wgen_x)
__global__ __launch_bounds__(128) void gemm_pgen_kernel(
    const bf16* __restrict__ decb, const bf16* __restrict__ h0b,
    const bf16* __restrict__ Wgx, const bf16* __restrict__ Wgh,
    const float* __restrict__ Zenc, float* __restrict__ pgen_s) {
  int wave = threadIdx.x >> 5, lane = threadIdx.x & 31;
  int tile = blockIdx.x * 4 + wave;   // 128*32 tiles
  int mt = tile >> 5, nt = tile & 31;
  v8f acc = {0.f, 0.f, 0.f, 0.f, 0.f, 0.f, 0.f, 0.f};
  for (int kt = 0; kt < 8; ++kt) {          // dec_in part, K = 256
    v16bf A = load_a_tile(decb, mt * 16, E_, kt * 32);
    v16bf Bm = load_b_tile(Wgx, kt * 32 + nt);
    acc = wmma_bf16(A, Bm, acc);
  }
  int b = mt >> 1;                           // 16-row tile lies in one batch row
  for (int kt = 0; kt < 16; ++kt) {         // h0 (pre-update) part, K = 512
    v16bf A = load_a_bcast(h0b + b * U_, kt * 32);
    v16bf Bm = load_b_tile(Wgh, kt * 32 + nt);
    acc = wmma_bf16(A, Bm, acc);
  }
  int col = nt * 16 + (lane & 15);
  int roff = (lane >> 4) << 3;
  for (int v = 0; v < 8; ++v) {
    int r = mt * 16 + v + roff;
    pgen_s[r * U_ + col] = sigm(acc[v] + Zenc[r * U_ + col]);
  }
}

// ---------------------------------------------------------------------------
// Persistent LSTM recurrence across NWG=8 workgroups (one u-slice of 64 each).
// h is double-buffered in global memory (L2-resident, 64KB bf16 per buffer);
// c stays in each WG's LDS (64 x 64 f32 = 16KB). One device barrier/timestep.
// Per wave/timestep: 16 kt x 4 gates = 64 WMMAs (4 accumulator tiles live).
// ---------------------------------------------------------------------------
__global__ __launch_bounds__(512) void lstm_seq_kernel(
    const float* __restrict__ xg, const bf16* __restrict__ Wrp,
    float* __restrict__ h_cur, float* __restrict__ c_cur,
    bf16* __restrict__ h0b, bf16* __restrict__ hg0, bf16* __restrict__ hg1,
    int* barrier_ctr, float* __restrict__ dec_outs_s,
    float* __restrict__ hs_s, float* __restrict__ cs_s) {
  __shared__ float c_sh[B_ * 64];                 // this WG's c slice (16 KB)
  int tid = threadIdx.x, wave = tid >> 5, lane = tid & 31;
  int wg = blockIdx.x;                            // 0..NWG-1
  int u0 = wg * 64;                               // owned u range [u0, u0+64)

  // init: our slice of h/c into hg0 / LDS, plus the step-start scan outputs
  for (int i = tid; i < B_ * 64; i += 512) {
    int b = i >> 6, uu = i & 63;
    int g = b * U_ + u0 + uu;
    float hv = h_cur[g], cv = c_cur[g];
    c_sh[i] = cv;
    hg0[g] = (bf16)hv;
    h0b[g] = (bf16)hv;     // pre-update h for the p_gen GEMM
    hs_s[g] = hv;
    cs_s[g] = cv;
  }
  grid_barrier(barrier_ctr, NWG);                 // everyone's h published

  int mt = wave >> 2;                             // batch tile 0..3
  int nt = wg * 4 + (wave & 3);                   // u-tile 0..31
  for (int t = 0; t < T_; ++t) {
    const bf16* hA = (t & 1) ? hg1 : hg0;
    bf16* hN = (t & 1) ? hg0 : hg1;
    v8f acc[4];
    for (int g = 0; g < 4; ++g)
      acc[g] = (v8f){0.f, 0.f, 0.f, 0.f, 0.f, 0.f, 0.f, 0.f};
    for (int kt = 0; kt < 16; ++kt) {
      v16bf A = load_a_tile(hA, mt * 16, U_, kt * 32);
      for (int g = 0; g < 4; ++g) {
        v16bf Bm = load_b_tile(Wrp, kt * 128 + g * 32 + nt);
        acc[g] = wmma_bf16(A, Bm, acc[g]);
      }
    }
    int u = nt * 16 + (lane & 15);
    int roff = (lane >> 4) << 3;
    for (int v = 0; v < 8; ++v) {
      int b = mt * 16 + v + roff;
      const float* xr = xg + (b * T_ + t) * G4;
      float gi = acc[0][v] + xr[u];
      float gf = acc[1][v] + xr[512 + u];
      float gc = acc[2][v] + xr[1024 + u];
      float go = acc[3][v] + xr[1536 + u];
      int ci = b * 64 + (u - u0);
      float cold = c_sh[ci];
      float cn = sigm(gf) * cold + sigm(gi) * tanhf(gc);
      float hn = sigm(go) * tanhf(cn);
      c_sh[ci] = cn;
      hN[b * U_ + u] = (bf16)hn;
      dec_outs_s[(b * T_ + t) * U_ + u] = hn;
      if (t == T_ - 1) {                          // carries for next outer step
        h_cur[b * U_ + u] = hn;
        c_cur[b * U_ + u] = cn;
      }
    }
    grid_barrier(barrier_ctr, (t + 2) * NWG);
  }
}

// argmax over U per row (first occurrence, like jnp.argmax) + embedding gather
__global__ __launch_bounds__(128) void argmax_embed_kernel(
    const float* __restrict__ outs_s, const float* __restrict__ emb,
    bf16* __restrict__ decb) {
  int wave = threadIdx.x >> 5, lane = threadIdx.x & 31;
  int row = blockIdx.x * 4 + wave;
  const float* p = outs_s + row * U_ + lane * 16;
  float mv = p[0]; int mi = lane * 16;
  for (int j = 1; j < 16; ++j) {
    float v = p[j];
    if (v > mv) { mv = v; mi = lane * 16 + j; }
  }
  for (int off = 16; off >= 1; off >>= 1) {
    float ov = __shfl_xor(mv, off, 32);
    int oi = __shfl_xor(mi, off, 32);
    if (ov > mv || (ov == mv && oi < mi)) { mv = ov; mi = oi; }
  }
  const float* e = emb + (long)mi * E_;
  for (int j = lane; j < E_; j += 32) decb[row * E_ + j] = (bf16)e[j];
}

// ---------------------------------------------------------------------------
extern "C" void kernel_launch(void* const* d_in, const int* in_sizes, int n_in,
                              void* d_out, int out_size, void* d_ws, size_t ws_size,
                              hipStream_t stream) {
  (void)in_sizes; (void)n_in; (void)out_size; (void)ws_size;
  const int*   target  = (const int*)d_in[1];
  const float* enc_out = (const float*)d_in[2];
  const float* enc_h   = (const float*)d_in[3];
  const float* enc_c   = (const float*)d_in[4];
  const float* emb     = (const float*)d_in[5];
  const float* Wk      = (const float*)d_in[6];
  const float* Wr      = (const float*)d_in[7];
  const float* bias    = (const float*)d_in[8];
  const float* Wgen    = (const float*)d_in[9];   // (1280, 512): enc|h|dec rows
  const float* bgen    = (const float*)d_in[10];

  // Workspace carve-up
  char* ws = (char*)d_ws;
  size_t off = 0;
  auto take = [&](size_t bytes) { char* p = ws + off; off += (bytes + 255) & ~(size_t)255; return p; };
  bf16*  Wkp  = (bf16*)take((size_t)E_ * G4 * 2);        // 1 MB
  bf16*  Wrp  = (bf16*)take((size_t)U_ * G4 * 2);        // 2 MB
  bf16*  Wge  = (bf16*)take((size_t)U_ * U_ * 2);        // 512 KB
  bf16*  Wgh  = (bf16*)take((size_t)U_ * U_ * 2);        // 512 KB
  bf16*  Wgx  = (bf16*)take((size_t)E_ * U_ * 2);        // 256 KB
  bf16*  encb = (bf16*)take((size_t)ROWS * U_ * 2);      // 2 MB
  float* Zenc = (float*)take((size_t)ROWS * U_ * 4);     // 4 MB
  float* xg   = (float*)take((size_t)ROWS * G4 * 4);     // 16 MB
  bf16*  decb = (bf16*)take((size_t)ROWS * E_ * 2);      // 1 MB
  float* hcur = (float*)take((size_t)B_ * U_ * 4);
  float* ccur = (float*)take((size_t)B_ * U_ * 4);
  bf16*  h0b  = (bf16*)take((size_t)B_ * U_ * 2);
  bf16*  hg0  = (bf16*)take((size_t)B_ * U_ * 2);        // h double buffer 0
  bf16*  hg1  = (bf16*)take((size_t)B_ * U_ * 2);        // h double buffer 1
  int*   ctr  = (int*)take(256);                          // device barrier ctr

  float* out      = (float*)d_out;
  float* dec_outs = out;                                   // 31*2048*512
  float* p_gens   = out + (size_t)S_ * ROWS * U_;
  float* hs       = p_gens + (size_t)S_ * ROWS * U_;       // 31*64*512
  float* cs       = hs + (size_t)S_ * B_ * U_;

  // --- one-time prep ---
  {
    int tot;
    tot = E_ * G4;  pack_b_kernel<<<(tot + 255) / 256, 256, 0, stream>>>(Wk, Wkp, E_, G4, tot);
    tot = U_ * G4;  pack_b_kernel<<<(tot + 255) / 256, 256, 0, stream>>>(Wr, Wrp, U_, G4, tot);
    tot = U_ * U_;  pack_b_kernel<<<(tot + 255) / 256, 256, 0, stream>>>(Wgen, Wge, U_, U_, tot);
    tot = U_ * U_;  pack_b_kernel<<<(tot + 255) / 256, 256, 0, stream>>>(Wgen + (size_t)U_ * U_, Wgh, U_, U_, tot);
    tot = E_ * U_;  pack_b_kernel<<<(tot + 255) / 256, 256, 0, stream>>>(Wgen + (size_t)2 * U_ * U_, Wgx, E_, U_, tot);
    tot = ROWS * U_;
    f32_to_bf16_kernel<<<(tot + 255) / 256, 256, 0, stream>>>(enc_out, encb, tot);
    init_carry_kernel<<<(B_ * U_ + 255) / 256, 256, 0, stream>>>(enc_h, enc_c, hcur, ccur, B_ * U_);
    embed_init_kernel<<<ROWS / 4, 128, 0, stream>>>(target, emb, decb);
    gemm_zenc_kernel<<<(128 * 32) / 4, 128, 0, stream>>>(encb, Wge, bgen, Zenc);
  }

  // --- 31 sequential outer steps ---
  for (int s = 0; s < S_; ++s) {
    float* dec_outs_s = dec_outs + (size_t)s * ROWS * U_;
    float* p_gens_s   = p_gens + (size_t)s * ROWS * U_;
    float* hs_s       = hs + (size_t)s * B_ * U_;
    float* cs_s       = cs + (size_t)s * B_ * U_;

    gemm_xg_kernel<<<(128 * 128) / 4, 128, 0, stream>>>(decb, Wkp, bias, xg, ctr);
    lstm_seq_kernel<<<NWG, 512, 0, stream>>>(xg, Wrp, hcur, ccur, h0b, hg0, hg1,
                                             ctr, dec_outs_s, hs_s, cs_s);
    gemm_pgen_kernel<<<(128 * 32) / 4, 128, 0, stream>>>(decb, h0b, Wgx, Wgh,
                                                         Zenc, p_gens_s);
    if (s < S_ - 1)
      argmax_embed_kernel<<<ROWS / 4, 128, 0, stream>>>(dec_outs_s, emb, decb);
  }
}
